// Gemma3Model_36043365548530
// MI455X (gfx1250) — compile-verified, matrix operand order checked by reading, not run on hardware
//
#include <hip/hip_runtime.h>
#include <hip/hip_bf16.h>
#include <math.h>

// ---------------- model constants ----------------
#define LAYERS 6
#define BB 2
#define SS 1024
#define VV 32000
#define EE 1024
#define HH 16
#define GG 4
#define HD 64
#define FF 4096
#define BS (BB * SS)   // 2048

typedef __attribute__((ext_vector_type(16))) __bf16 v16bf;
typedef __attribute__((ext_vector_type(8)))  float  v8f;
typedef unsigned int u32;

__device__ __forceinline__ __bf16 f2bf(float f) {
    union { float f; unsigned u; } in; in.f = f;
    unsigned r = in.u + 0x7FFFu + ((in.u >> 16) & 1u);   // round-to-nearest-even
    union { unsigned short s; __bf16 b; } out; out.s = (unsigned short)(r >> 16);
    return out.b;
}

// =====================================================================
// Tiled bf16 WMMA GEMM with register-staged software pipeline.
//   C[M,N] = A[M,K] @ W[K,N]  (+ optional addend)
// A, W bf16 in global. BM=128, BN=128, BK=32; 256 threads = 8 waves in a
// 4x2 grid; each wave computes 32x64 via 2 A-frags x 4 B-frags =
// 8 v_wmma_f32_16x16x32_bf16 per K-step, f32 accumulate.
// Pipeline: tile t+1 global loads issue before the WMMA burst of tile t.
// =====================================================================
#define BM 128
#define BN 128
#define BK 32
#define SPITCH 48    // padded LDS row pitch in bf16 (96B, 16B aligned)

__global__ __launch_bounds__(256) void gemm_bf16_wmma(
    const __bf16* __restrict__ A, const __bf16* __restrict__ W,
    float* __restrict__ C, const float* __restrict__ addend,
    int M, int N, int K)
{
    __shared__ __bf16 sA[BM * SPITCH];    // [row][k]
    __shared__ __bf16 sBt[BN * SPITCH];   // transposed: [n][k]

    const int tid  = threadIdx.x;
    const int wave = tid >> 5;
    const int lane = tid & 31;
    const int lr   = lane & 15;
    const int kh   = lane >> 4;          // K-half select per ISA bf16 layout
    const int wm   = wave >> 1;          // 0..3  (row group of 32)
    const int wn   = wave & 1;           // 0..1  (col group of 64)
    const int m0   = blockIdx.y * BM;
    const int n0   = blockIdx.x * BN;

    v8f acc[2][4];
    #pragma unroll
    for (int mi = 0; mi < 2; ++mi)
        #pragma unroll
        for (int nt = 0; nt < 4; ++nt)
            #pragma unroll
            for (int r = 0; r < 8; ++r) acc[mi][nt][r] = 0.0f;

    // A staging: thread copies 16 bf16 (2 x uint4): row = tid>>1, col half
    const int ar = tid >> 1;
    const int ac = (tid & 1) * 16;
    // B staging: thread handles k-pair (tid>>4)*2 and 8 n values (tid&15)*8
    const int bkp = (tid >> 4) * 2;
    const int bn  = (tid & 15) * 8;

    const __bf16* aBase = A + (size_t)(m0 + ar) * K + ac;           // + k0
    const __bf16* bBase = W + (size_t)bkp * N + n0 + bn;            // + k0*N

    const int nTiles = K / BK;

    // ---- preload tile 0 into registers ----
    uint4 aR0, aR1, bR0, bR1;
    {
        const uint4* ap = (const uint4*)(const void*)(aBase);
        aR0 = ap[0]; aR1 = ap[1];
        const uint4* bp = (const uint4*)(const void*)(bBase);
        bR0 = *bp;
        bR1 = *(const uint4*)(const void*)(bBase + N);
    }

    for (int t = 0; t < nTiles; ++t) {
        // ---- store staged registers to LDS ----
        *(uint4*)(void*)&sA[ar * SPITCH + ac]     = aR0;
        *(uint4*)(void*)&sA[ar * SPITCH + ac + 8] = aR1;
        {
            u32 pa[4] = {bR0.x, bR0.y, bR0.z, bR0.w};
            u32 pb[4] = {bR1.x, bR1.y, bR1.z, bR1.w};
            #pragma unroll
            for (int j = 0; j < 4; ++j) {
                u32 lo = (pa[j] & 0xffffu) | (pb[j] << 16);          // n=bn+2j
                u32 hi = (pa[j] >> 16)     | (pb[j] & 0xffff0000u);  // n=bn+2j+1
                *(u32*)(void*)&sBt[(bn + 2 * j)     * SPITCH + bkp] = lo;
                *(u32*)(void*)&sBt[(bn + 2 * j + 1) * SPITCH + bkp] = hi;
            }
        }
        __syncthreads();

        // ---- issue global loads for tile t+1 (hidden under WMMA burst) ----
        if (t + 1 < nTiles) {
            const __bf16* ap = aBase + (size_t)(t + 1) * BK;
            const uint4* apv = (const uint4*)(const void*)ap;
            aR0 = apv[0]; aR1 = apv[1];
            const __bf16* bp = bBase + (size_t)(t + 1) * BK * N;
            bR0 = *(const uint4*)(const void*)bp;
            bR1 = *(const uint4*)(const void*)(bp + N);
        }
        if (t + 2 < nTiles) {   // global_prefetch_b8 one more stage ahead
            __builtin_prefetch(aBase + (size_t)(t + 2) * BK, 0, 1);
            __builtin_prefetch(bBase + (size_t)(t + 2) * BK * N, 0, 1);
        }

        // ---- build all fragments, then back-to-back WMMAs ----
        // elems 0-7 => K = kh*8+i, elems 8-15 => K = 16+kh*8+i
        v16bf af[2];
        #pragma unroll
        for (int mi = 0; mi < 2; ++mi) {
            const __bf16* base = &sA[(wm * 32 + mi * 16 + lr) * SPITCH];
            #pragma unroll
            for (int i = 0; i < 8; ++i) af[mi][i]     = base[kh * 8 + i];
            #pragma unroll
            for (int i = 0; i < 8; ++i) af[mi][8 + i] = base[16 + kh * 8 + i];
        }
        v16bf bfv[4];
        #pragma unroll
        for (int nt = 0; nt < 4; ++nt) {
            const __bf16* base = &sBt[(wn * 64 + nt * 16 + lr) * SPITCH];
            #pragma unroll
            for (int i = 0; i < 8; ++i) bfv[nt][i]     = base[kh * 8 + i];
            #pragma unroll
            for (int i = 0; i < 8; ++i) bfv[nt][8 + i] = base[16 + kh * 8 + i];
        }
        #pragma unroll
        for (int nt = 0; nt < 4; ++nt) {
            acc[0][nt] = __builtin_amdgcn_wmma_f32_16x16x32_bf16(
                false, af[0], false, bfv[nt], (short)0, acc[0][nt], false, false);
            acc[1][nt] = __builtin_amdgcn_wmma_f32_16x16x32_bf16(
                false, af[1], false, bfv[nt], (short)0, acc[1][nt], false, false);
        }
        __syncthreads();
    }

    // C/D layout: lanes 0-15 => M=r, lanes 16-31 => M=8+r; N = lr
    #pragma unroll
    for (int mi = 0; mi < 2; ++mi) {
        const int rowb = m0 + wm * 32 + mi * 16 + kh * 8;
        #pragma unroll
        for (int nt = 0; nt < 4; ++nt) {
            const int col = n0 + wn * 64 + nt * 16 + lr;
            #pragma unroll
            for (int r = 0; r < 8; ++r) {
                const size_t idx = (size_t)(rowb + r) * N + col;
                float v = acc[mi][nt][r];
                if (addend) v += addend[idx];
                C[idx] = v;
            }
        }
    }
}

// =====================================================================
// Elementwise / reduction kernels
// =====================================================================
__global__ void f32_to_bf16_kernel(const float* __restrict__ src,
                                   __bf16* __restrict__ dst, int n)
{
    int i = blockIdx.x * blockDim.x + threadIdx.x;
    if (i < n) dst[i] = f2bf(src[i]);
}

__global__ void embed_kernel(const int* __restrict__ ids,
                             const float* __restrict__ emb,
                             float* __restrict__ x, int total)
{
    int i = blockIdx.x * blockDim.x + threadIdx.x;
    if (i < total) {
        int tok = ids[i / EE];
        x[i] = emb[(size_t)tok * EE + (i % EE)];
    }
}

// RMSNorm, bf16 output (feeds WMMA GEMMs directly)
__global__ __launch_bounds__(256) void rmsnorm_bf16_kernel(
    const float* __restrict__ x, const float* __restrict__ g,
    __bf16* __restrict__ out)
{
    const int row = blockIdx.x;
    const float* xr = x + (size_t)row * EE;
    __shared__ float red[256];
    float s = 0.0f;
    for (int i = threadIdx.x; i < EE; i += 256) { float v = xr[i]; s += v * v; }
    red[threadIdx.x] = s; __syncthreads();
    for (int st = 128; st > 0; st >>= 1) {
        if (threadIdx.x < st) red[threadIdx.x] += red[threadIdx.x + st];
        __syncthreads();
    }
    const float rinv = rsqrtf(red[0] / (float)EE + 1e-6f);
    for (int i = threadIdx.x; i < EE; i += 256)
        out[(size_t)row * EE + i] = f2bf(xr[i] * rinv * g[i]);
}

// layout [bs][nh][64]; rotate-half RoPE, optional post-scale (for q)
__global__ void rope_kernel(float* __restrict__ q, int nh, float scale, int total)
{
    int i = blockIdx.x * blockDim.x + threadIdx.x;
    if (i >= total) return;
    const int d  = i & 31;
    const int h  = (i >> 5) % nh;
    const int bs = i / (32 * nh);
    const int s  = bs % SS;
    const float inv = __powf(10000.0f, -2.0f * (float)d / (float)HD);
    const float ang = (float)s * inv;
    const float c = __cosf(ang), sn = __sinf(ang);
    const size_t base = ((size_t)bs * nh + h) * HD;
    const float x1 = q[base + d], x2 = q[base + d + 32];
    q[base + d]      = (x1 * c - x2 * sn) * scale;
    q[base + d + 32] = (x2 * c + x1 * sn) * scale;
}

// grid (S, H, B); block 128; causal attention with GQA, two-pass softmax;
// writes bf16 ctx (input to the Wo WMMA GEMM)
__global__ __launch_bounds__(128) void attn_kernel(
    const float* __restrict__ q, const float* __restrict__ k,
    const float* __restrict__ v, __bf16* __restrict__ ctx)
{
    const int s = blockIdx.x, h = blockIdx.y, b = blockIdx.z;
    const int g = h / (HH / GG);
    __shared__ float p[SS];
    __shared__ float red[128];
    const float* qp = q + (((size_t)b * SS + s) * HH + h) * HD;

    float lmax = -1e30f;
    for (int t = threadIdx.x; t <= s; t += 128) {
        const float* kp = k + (((size_t)b * SS + t) * GG + g) * HD;
        float d = 0.0f;
        #pragma unroll 8
        for (int e = 0; e < HD; ++e) d += qp[e] * kp[e];
        p[t] = d;
        lmax = fmaxf(lmax, d);
    }
    red[threadIdx.x] = lmax; __syncthreads();
    for (int st = 64; st > 0; st >>= 1) {
        if (threadIdx.x < st) red[threadIdx.x] = fmaxf(red[threadIdx.x], red[threadIdx.x + st]);
        __syncthreads();
    }
    const float m = red[0]; __syncthreads();

    float lsum = 0.0f;
    for (int t = threadIdx.x; t <= s; t += 128) {
        const float e = __expf(p[t] - m);
        p[t] = e; lsum += e;
    }
    red[threadIdx.x] = lsum; __syncthreads();
    for (int st = 64; st > 0; st >>= 1) {
        if (threadIdx.x < st) red[threadIdx.x] += red[threadIdx.x + st];
        __syncthreads();
    }
    const float inv = 1.0f / red[0];

    if (threadIdx.x < HD) {
        const int e = threadIdx.x;
        float a = 0.0f;
        for (int t = 0; t <= s; ++t)
            a += p[t] * v[(((size_t)b * SS + t) * GG + g) * HD + e];
        ctx[(((size_t)b * SS + s) * HH + h) * HD + e] = f2bf(a * inv);
    }
}

__global__ void gelu_gate_kernel(const float* __restrict__ a,
                                 const float* __restrict__ g,
                                 __bf16* __restrict__ o, int total)
{
    int i = blockIdx.x * blockDim.x + threadIdx.x;
    if (i >= total) return;
    const float x = a[i];
    const float t = tanhf(0.7978845608028654f * (x + 0.044715f * x * x * x));
    o[i] = f2bf(0.5f * x * (1.0f + t) * g[i]);
}

__global__ void zero_kernel(float* p) { p[0] = 0.0f; }

__global__ __launch_bounds__(256) void nll_kernel(
    const float* __restrict__ logits, const int* __restrict__ targets,
    float* __restrict__ acc)
{
    const int row = blockIdx.x;
    const float* lr = logits + (size_t)row * VV;
    __shared__ float red[256];
    float m = -1e30f;
    for (int i = threadIdx.x; i < VV; i += 256) m = fmaxf(m, lr[i]);
    red[threadIdx.x] = m; __syncthreads();
    for (int st = 128; st > 0; st >>= 1) {
        if (threadIdx.x < st) red[threadIdx.x] = fmaxf(red[threadIdx.x], red[threadIdx.x + st]);
        __syncthreads();
    }
    m = red[0]; __syncthreads();
    float s = 0.0f;
    for (int i = threadIdx.x; i < VV; i += 256) s += __expf(lr[i] - m);
    red[threadIdx.x] = s; __syncthreads();
    for (int st = 128; st > 0; st >>= 1) {
        if (threadIdx.x < st) red[threadIdx.x] += red[threadIdx.x + st];
        __syncthreads();
    }
    if (threadIdx.x == 0) {
        const int t = targets[row];
        const float nll = -(lr[t] - m - __logf(red[0]));
        atomicAdd(acc, nll);
    }
}

__global__ void loss_finalize_kernel(const float* __restrict__ acc,
                                     float* __restrict__ out, float inv_n)
{
    out[0] = acc[0] * inv_n;
}

// =====================================================================
// Orchestration
// =====================================================================
extern "C" void kernel_launch(void* const* d_in, const int* in_sizes, int n_in,
                              void* d_out, int out_size, void* d_ws, size_t ws_size,
                              hipStream_t stream)
{
    const int*   input_ids = (const int*)  d_in[0];
    const int*   targets   = (const int*)  d_in[1];
    const float* tok_emb   = (const float*)d_in[2];
    const float* Wq        = (const float*)d_in[3];
    const float* Wk        = (const float*)d_in[4];
    const float* Wv        = (const float*)d_in[5];
    const float* Wo        = (const float*)d_in[6];
    const float* W1        = (const float*)d_in[7];
    const float* Wg        = (const float*)d_in[8];
    const float* W3        = (const float*)d_in[9];
    const float* g1        = (const float*)d_in[10];
    const float* g2        = (const float*)d_in[11];
    const float* gf        = (const float*)d_in[12];
    const float* Wout      = (const float*)d_in[13];

    float* logits = (float*)d_out;                      // [BS, V]
    float* lossp  = logits + (size_t)BS * VV;           // scalar loss

    // ---------------- workspace layout ----------------
    // fp32 region
    float* x    = (float*)d_ws;                      // BS*E
    float* q    = x    + (size_t)BS * EE;            // BS*H*HD
    float* kbuf = q    + (size_t)BS * HH * HD;       // BS*G*HD
    float* vbuf = kbuf + (size_t)BS * GG * HD;       // BS*G*HD
    float* ff1  = vbuf + (size_t)BS * GG * HD;       // BS*FF
    float* ffg  = ff1  + (size_t)BS * FF;            // BS*FF
    float* acc  = ffg  + (size_t)BS * FF;            // 1 (+pad)
    // bf16 region
    __bf16* hb    = (__bf16*)(acc + 8);              // BS*E
    __bf16* ctxb  = hb    + (size_t)BS * EE;         // BS*H*HD
    __bf16* ffhb  = ctxb  + (size_t)BS * HH * HD;    // BS*FF
    __bf16* wqb   = ffhb  + (size_t)BS * FF;         // 6*E*H*HD
    __bf16* wkb   = wqb   + (size_t)LAYERS * EE * HH * HD;
    __bf16* wvb   = wkb   + (size_t)LAYERS * EE * GG * HD;
    __bf16* wob   = wvb   + (size_t)LAYERS * EE * GG * HD;
    __bf16* w1b   = wob   + (size_t)LAYERS * HH * HD * EE;
    __bf16* wgb   = w1b   + (size_t)LAYERS * EE * FF;
    __bf16* w3b   = wgb   + (size_t)LAYERS * EE * FF;
    __bf16* woutb = w3b   + (size_t)LAYERS * FF * EE;

    auto cvt = [&](const float* src, __bf16* dst, long long n) {
        f32_to_bf16_kernel<<<(int)((n + 255) / 256), 256, 0, stream>>>(src, dst, (int)n);
    };
    auto gemm = [&](const __bf16* A, const __bf16* W, float* C, const float* add,
                    int M, int N, int K) {
        dim3 grid(N / BN, M / BM);
        gemm_bf16_wmma<<<grid, dim3(256), 0, stream>>>(A, W, C, add, M, N, K);
    };

    // one-shot weight conversion fp32 -> bf16 (GEMMs are compute-bound;
    // this keeps the WMMA hot loop a pure byte copy)
    cvt(Wq,   wqb,   (long long)LAYERS * EE * HH * HD);
    cvt(Wk,   wkb,   (long long)LAYERS * EE * GG * HD);
    cvt(Wv,   wvb,   (long long)LAYERS * EE * GG * HD);
    cvt(Wo,   wob,   (long long)LAYERS * HH * HD * EE);
    cvt(W1,   w1b,   (long long)LAYERS * EE * FF);
    cvt(Wg,   wgb,   (long long)LAYERS * EE * FF);
    cvt(W3,   w3b,   (long long)LAYERS * FF * EE);
    cvt(Wout, woutb, (long long)EE * VV);

    // embedding gather
    {
        const int total = BS * EE;
        embed_kernel<<<(total + 255) / 256, 256, 0, stream>>>(input_ids, tok_emb, x, total);
    }

    const float scale = 1.0f / sqrtf((float)HD);

    for (int l = 0; l < LAYERS; ++l) {
        const __bf16* wq = wqb + (size_t)l * EE * HH * HD;
        const __bf16* wk = wkb + (size_t)l * EE * GG * HD;
        const __bf16* wv = wvb + (size_t)l * EE * GG * HD;
        const __bf16* wo = wob + (size_t)l * HH * HD * EE;
        const __bf16* w1 = w1b + (size_t)l * EE * FF;
        const __bf16* wg = wgb + (size_t)l * EE * FF;
        const __bf16* w3 = w3b + (size_t)l * FF * EE;

        // attention block
        rmsnorm_bf16_kernel<<<BS, 256, 0, stream>>>(x, g1 + (size_t)l * EE, hb);
        gemm(hb, wq, q,    nullptr, BS, HH * HD, EE);
        gemm(hb, wk, kbuf, nullptr, BS, GG * HD, EE);
        gemm(hb, wv, vbuf, nullptr, BS, GG * HD, EE);
        {
            int tq = BS * HH * 32;
            rope_kernel<<<(tq + 255) / 256, 256, 0, stream>>>(q, HH, scale, tq);
            int tk = BS * GG * 32;
            rope_kernel<<<(tk + 255) / 256, 256, 0, stream>>>(kbuf, GG, 1.0f, tk);
        }
        attn_kernel<<<dim3(SS, HH, BB), 128, 0, stream>>>(q, kbuf, vbuf, ctxb);
        gemm(ctxb, wo, x, x, BS, EE, HH * HD);   // x = x + ctx @ Wo

        // FFN block
        rmsnorm_bf16_kernel<<<BS, 256, 0, stream>>>(x, g2 + (size_t)l * EE, hb);
        gemm(hb, w1, ff1, nullptr, BS, FF, EE);
        gemm(hb, wg, ffg, nullptr, BS, FF, EE);
        {
            const int total = BS * FF;
            gelu_gate_kernel<<<(total + 255) / 256, 256, 0, stream>>>(ff1, ffg, ffhb, total);
        }
        gemm(ffhb, w3, x, x, BS, EE, FF);        // x = x + ffh @ W3
    }

    // final norm + vocab projection + loss
    rmsnorm_bf16_kernel<<<BS, 256, 0, stream>>>(x, gf, hb);
    gemm(hb, woutb, logits, nullptr, BS, VV, EE);

    zero_kernel<<<1, 1, 0, stream>>>(acc);
    nll_kernel<<<BS, 256, 0, stream>>>(logits, targets, acc);
    loss_finalize_kernel<<<1, 1, 0, stream>>>(acc, lossp, 1.0f / (float)BS);
}
